// MultiheadAttention_7997229105671
// MI455X (gfx1250) — compile-verified, matrix operand order checked by reading, not run on hardware
//
#include <hip/hip_runtime.h>
#include <cstddef>

// MI455X / gfx1250, wave32, WMMA bf16 path.
// B=4, S=2048, D=1024, H=16, DK=64.

typedef __attribute__((ext_vector_type(16))) __bf16 v16bf;
typedef __attribute__((ext_vector_type(8)))  float  v8f;
typedef __attribute__((ext_vector_type(4)))  unsigned int u32x4;
typedef __attribute__((ext_vector_type(4)))  float  f32x4;
typedef __attribute__((ext_vector_type(4)))  __bf16 v4bf;

#define S_LEN 2048
#define D_DIM 1024
#define H_NUM 16
#define DK_SZ 64
#define M_ROWS (4 * S_LEN)   // 8192

// ---------------------------------------------------------------------------
// f32 -> bf16 conversion (vectorized x4)
// ---------------------------------------------------------------------------
__global__ __launch_bounds__(256) void cvt_f32_bf16(const float* __restrict__ in,
                                                    __bf16* __restrict__ out, int n4) {
    int i = blockIdx.x * blockDim.x + threadIdx.x;
    if (i < n4) {
        f32x4 v = ((const f32x4*)in)[i];
        v4bf o;
        o.x = (__bf16)v.x; o.y = (__bf16)v.y; o.z = (__bf16)v.z; o.w = (__bf16)v.w;
        ((v4bf*)out)[i] = o;
    }
}

// ---------------------------------------------------------------------------
// GEMM: C[m,n] = sum_k A[m,k] * W[n,k]   (both operands K-contiguous)
// Block tile 128x128, 8 waves (2x4), wave tile 64x32 = 4x2 wmma tiles, K step 32.
// mode 0: out bf16 [B,H,S,DK]  (Q,K)
// mode 1: out bf16 [B,H,DK,S]  (V transposed)
// mode 2: out f32  [M,N] row-major (final projection -> d_out)
// ---------------------------------------------------------------------------
__global__ __launch_bounds__(256) void wmma_gemm(const __bf16* __restrict__ A,
                                                 const __bf16* __restrict__ W,
                                                 void* __restrict__ Cout,
                                                 int Kd, int mode) {
    __shared__ __bf16 As[128][32];   // 8 KB
    __shared__ __bf16 Bs[128][32];   // 8 KB

    const int tid   = threadIdx.x;
    const int lane  = tid & 31;
    const int wave  = tid >> 5;      // 0..7
    const int waveM = wave >> 2;     // 0..1  (64 rows each)
    const int waveN = wave & 3;      // 0..3  (32 cols each)
    const int bm    = blockIdx.x * 128;
    const int bn    = blockIdx.y * 128;

    const int lrow = lane & 15;
    const int lhi  = lane >> 4;
    const int kb   = lhi * 8;

    v8f zero = {};
    v8f acc[4][2];
    for (int mt = 0; mt < 4; ++mt)
        for (int nt = 0; nt < 2; ++nt) acc[mt][nt] = zero;

    for (int k0 = 0; k0 < Kd; k0 += 32) {
        // Cooperative tile load: 128x32 bf16 = 512 16-byte chunks per matrix.
        for (int c = 0; c < 2; ++c) {
            int chunk = tid + c * 256;           // 0..511
            int r  = chunk >> 2;                 // row 0..127
            int cc = (chunk & 3) * 8;            // col (bf16 units)
            *(u32x4*)&As[r][cc] = *(const u32x4*)&A[(size_t)(bm + r) * Kd + k0 + cc];
            *(u32x4*)&Bs[r][cc] = *(const u32x4*)&W[(size_t)(bn + r) * Kd + k0 + cc];
        }
        __syncthreads();

        v16bf af[4], bfrag[2];
        for (int mt = 0; mt < 4; ++mt) {
            int r = waveM * 64 + mt * 16 + lrow;
            union { u32x4 u[2]; v16bf v; } f;
            f.u[0] = *(const u32x4*)&As[r][kb];
            f.u[1] = *(const u32x4*)&As[r][kb + 16];
            af[mt] = f.v;
        }
        for (int nt = 0; nt < 2; ++nt) {
            int r = waveN * 32 + nt * 16 + lrow;
            union { u32x4 u[2]; v16bf v; } f;
            f.u[0] = *(const u32x4*)&Bs[r][kb];
            f.u[1] = *(const u32x4*)&Bs[r][kb + 16];
            bfrag[nt] = f.v;
        }
        for (int mt = 0; mt < 4; ++mt)
            for (int nt = 0; nt < 2; ++nt)
                acc[mt][nt] = __builtin_amdgcn_wmma_f32_16x16x32_bf16(
                    false, af[mt], false, bfrag[nt], (short)0, acc[mt][nt], false, false);
        __syncthreads();
    }

    // Store. C layout: VGPR g -> row g + 8*lhi, col lrow.
    for (int mt = 0; mt < 4; ++mt) {
        for (int nt = 0; nt < 2; ++nt) {
            for (int g = 0; g < 8; ++g) {
                int m = bm + waveM * 64 + mt * 16 + g + 8 * lhi;
                int n = bn + waveN * 32 + nt * 16 + lrow;
                float v = acc[mt][nt][g];
                if (mode == 2) {
                    ((float*)Cout)[(size_t)m * D_DIM + n] = v;
                } else {
                    __bf16 bv = (__bf16)v;
                    int b = m >> 11, s = m & (S_LEN - 1);
                    int h = n >> 6,  dk = n & (DK_SZ - 1);
                    if (mode == 0)
                        ((__bf16*)Cout)[(((size_t)b * H_NUM + h) * S_LEN + s) * DK_SZ + dk] = bv;
                    else
                        ((__bf16*)Cout)[(((size_t)b * H_NUM + h) * DK_SZ + dk) * S_LEN + s] = bv;
                }
            }
        }
    }
}

// ---------------------------------------------------------------------------
// Flash attention: per block (qt, h, b), 4 waves x 16 query rows = 64 queries.
// Q,K: [B,H,S,64] bf16 ; VT: [B,H,64,S] bf16 ; out: bf16 [B*S, 1024].
// Online softmax in f32 accumulators (C layout), P staged via per-wave LDS.
// ---------------------------------------------------------------------------
__global__ __launch_bounds__(128) void flash_attn(const __bf16* __restrict__ Q,
                                                  const __bf16* __restrict__ K,
                                                  const __bf16* __restrict__ VT,
                                                  __bf16* __restrict__ Oa,
                                                  const int* __restrict__ causal_p) {
    __shared__ __bf16 Ks[64][64];       // [key][dk]   8 KB
    __shared__ __bf16 Vs[64][64];       // [dk][key]   8 KB
    __shared__ __bf16 Ps[4][16][64];    // per-wave P  8 KB

    const int tid  = threadIdx.x;
    const int lane = tid & 31;
    const int wave = tid >> 5;          // 0..3
    const int lrow = lane & 15;
    const int lhi  = lane >> 4;
    const int kb   = lhi * 8;

    const int qt = blockIdx.x;          // 0..31
    const int h  = blockIdx.y;
    const int b  = blockIdx.z;
    const int causal = *causal_p;

    const size_t headoff = ((size_t)b * H_NUM + h) * S_LEN * DK_SZ;
    const __bf16* Qh = Q  + headoff;
    const __bf16* Kh = K  + headoff;
    const __bf16* Vh = VT + headoff;

    const int qbase = qt * 64;
    const int qfrow = qbase + wave * 16 + lrow;

    // Q A-fragments, reused across all key tiles.
    v16bf qf[2];
    for (int ks = 0; ks < 2; ++ks) {
        union { u32x4 u[2]; v16bf v; } f;
        const __bf16* p = Qh + (size_t)qfrow * DK_SZ + ks * 32 + kb;
        f.u[0] = *(const u32x4*)p;
        f.u[1] = *(const u32x4*)(p + 16);
        qf[ks] = f.v;
    }

    v8f zero = {};
    v8f oacc[4];
    for (int nt = 0; nt < 4; ++nt) oacc[nt] = zero;
    float mrow[8], lsum[8];
    for (int g = 0; g < 8; ++g) { mrow[g] = -INFINITY; lsum[g] = 0.0f; }

    const int ktiles = causal ? (qt + 1) : (S_LEN / 64);

    for (int kt = 0; kt < ktiles; ++kt) {
        // Cooperative K/V tile load: 64x64 bf16 = 512 16-B chunks each, 4 per thread.
        for (int c = 0; c < 4; ++c) {
            int chunk = tid + c * 128;
            int r  = chunk >> 3;
            int cc = (chunk & 7) * 8;
            *(u32x4*)&Ks[r][cc] = *(const u32x4*)&Kh[((size_t)kt * 64 + r) * DK_SZ + cc];
            *(u32x4*)&Vs[r][cc] = *(const u32x4*)&Vh[(size_t)r * S_LEN + kt * 64 + cc];
        }
        if (kt + 1 < ktiles) {   // global_prefetch_b8 of next tiles
            __builtin_prefetch(&Kh[((size_t)(kt + 1) * 64) * DK_SZ + tid * 32], 0, 1);
            __builtin_prefetch(&Vh[(size_t)(tid >> 1) * S_LEN + (kt + 1) * 64 + (tid & 1) * 32], 0, 1);
        }
        __syncthreads();

        // Scores: S = Q (16x64) . K^T (64 keys), 4 N-subtiles, K=64 via 2 wmma.
        v8f sc[4];
        for (int nt = 0; nt < 4; ++nt) {
            v8f s = zero;
            for (int ks = 0; ks < 2; ++ks) {
                union { u32x4 u[2]; v16bf v; } f;
                const __bf16* p = &Ks[nt * 16 + lrow][ks * 32 + kb];
                f.u[0] = *(const u32x4*)p;
                f.u[1] = *(const u32x4*)(p + 16);
                s = __builtin_amdgcn_wmma_f32_16x16x32_bf16(
                        false, qf[ks], false, f.v, (short)0, s, false, false);
            }
            sc[nt] = s * 0.125f;    // 1/sqrt(64)
        }

        if (causal && kt == qt) {   // diagonal tile mask
            for (int nt = 0; nt < 4; ++nt) {
                int key = kt * 64 + nt * 16 + lrow;
                for (int g = 0; g < 8; ++g) {
                    int q = qbase + wave * 16 + g + 8 * lhi;
                    if (key > q) sc[nt][g] = -1e30f;
                }
            }
        }

        // Row max (per C-layout row = (g, half-wave)): reduce across 16 lanes.
        float alpha[8];
        for (int g = 0; g < 8; ++g) {
            float v = sc[0][g];
            v = fmaxf(v, sc[1][g]); v = fmaxf(v, sc[2][g]); v = fmaxf(v, sc[3][g]);
            v = fmaxf(v, __shfl_xor(v, 1, 32));
            v = fmaxf(v, __shfl_xor(v, 2, 32));
            v = fmaxf(v, __shfl_xor(v, 4, 32));
            v = fmaxf(v, __shfl_xor(v, 8, 32));
            float mnew = fmaxf(mrow[g], v);
            alpha[g] = __expf(mrow[g] - mnew);
            mrow[g] = mnew;
        }
        // P = exp(S - m), row sums, rescale running state.
        float rs[8];
        for (int g = 0; g < 8; ++g) rs[g] = 0.0f;
        for (int nt = 0; nt < 4; ++nt)
            for (int g = 0; g < 8; ++g) {
                float p = __expf(sc[nt][g] - mrow[g]);
                sc[nt][g] = p;
                rs[g] += p;
            }
        for (int g = 0; g < 8; ++g) {
            float v = rs[g];
            v += __shfl_xor(v, 1, 32);
            v += __shfl_xor(v, 2, 32);
            v += __shfl_xor(v, 4, 32);
            v += __shfl_xor(v, 8, 32);
            lsum[g] = lsum[g] * alpha[g] + v;
        }
        for (int nt = 0; nt < 4; ++nt)
            for (int g = 0; g < 8; ++g)
                oacc[nt][g] *= alpha[g];

        // C-layout -> A-layout for P via per-wave LDS staging.
        for (int nt = 0; nt < 4; ++nt)
            for (int g = 0; g < 8; ++g)
                Ps[wave][g + 8 * lhi][nt * 16 + lrow] = (__bf16)sc[nt][g];
        asm volatile("s_wait_dscnt 0" ::: "memory");   // wave-local RAW on LDS

        v16bf pf[2];
        for (int ks = 0; ks < 2; ++ks) {
            union { u32x4 u[2]; v16bf v; } f;
            const __bf16* p = &Ps[wave][lrow][ks * 32 + kb];
            f.u[0] = *(const u32x4*)p;
            f.u[1] = *(const u32x4*)(p + 16);
            pf[ks] = f.v;
        }
        // O += P (16x64) . V (64 x 64dk): B-operand from VT (key-contiguous).
        for (int nt = 0; nt < 4; ++nt) {
            for (int ks = 0; ks < 2; ++ks) {
                union { u32x4 u[2]; v16bf v; } f;
                const __bf16* p = &Vs[nt * 16 + lrow][ks * 32 + kb];
                f.u[0] = *(const u32x4*)p;
                f.u[1] = *(const u32x4*)(p + 16);
                oacc[nt] = __builtin_amdgcn_wmma_f32_16x16x32_bf16(
                               false, pf[ks], false, f.v, (short)0, oacc[nt], false, false);
            }
        }
        __syncthreads();
    }

    // Normalize and store attention output, row-major [B*S, 1024] bf16.
    for (int nt = 0; nt < 4; ++nt) {
        for (int g = 0; g < 8; ++g) {
            int q = qbase + wave * 16 + g + 8 * lhi;
            float inv = (lsum[g] > 0.0f) ? 1.0f / lsum[g] : 0.0f;
            float v = oacc[nt][g] * inv;
            Oa[((size_t)b * S_LEN + q) * D_DIM + h * DK_SZ + nt * 16 + lrow] = (__bf16)v;
        }
    }
}

// ---------------------------------------------------------------------------
// Host-side orchestration (graph-capture safe: launches only).
// ---------------------------------------------------------------------------
extern "C" void kernel_launch(void* const* d_in, const int* in_sizes, int n_in,
                              void* d_out, int out_size, void* d_ws, size_t ws_size,
                              hipStream_t stream) {
    (void)in_sizes; (void)n_in; (void)out_size; (void)ws_size;

    const float* x  = (const float*)d_in[0];
    const float* Wq = (const float*)d_in[1];
    const float* Wk = (const float*)d_in[2];
    const float* Wv = (const float*)d_in[3];
    const float* Wo = (const float*)d_in[4];
    const int* causal = (const int*)d_in[5];

    size_t off = 0;
    auto alloc = [&](size_t bytes) -> void* {
        void* p = (char*)d_ws + off;
        off += (bytes + 255) & ~(size_t)255;
        return p;
    };

    const size_t xN = (size_t)M_ROWS * D_DIM;        // 8,388,608
    const size_t wN = (size_t)D_DIM * D_DIM;         // 1,048,576

    __bf16* xb  = (__bf16*)alloc(xN * 2);
    __bf16* wqb = (__bf16*)alloc(wN * 2);
    __bf16* wkb = (__bf16*)alloc(wN * 2);
    __bf16* wvb = (__bf16*)alloc(wN * 2);
    __bf16* wob = (__bf16*)alloc(wN * 2);
    __bf16* Qb  = (__bf16*)alloc(xN * 2);
    __bf16* Kb  = (__bf16*)alloc(xN * 2);
    __bf16* VTb = (__bf16*)alloc(xN * 2);
    __bf16* At  = (__bf16*)alloc(xN * 2);

    // 1) convert inputs to bf16
    {
        int n4 = (int)(xN / 4);
        cvt_f32_bf16<<<(n4 + 255) / 256, 256, 0, stream>>>(x, xb, n4);
        int w4 = (int)(wN / 4);
        cvt_f32_bf16<<<(w4 + 255) / 256, 256, 0, stream>>>(Wq, wqb, w4);
        cvt_f32_bf16<<<(w4 + 255) / 256, 256, 0, stream>>>(Wk, wkb, w4);
        cvt_f32_bf16<<<(w4 + 255) / 256, 256, 0, stream>>>(Wv, wvb, w4);
        cvt_f32_bf16<<<(w4 + 255) / 256, 256, 0, stream>>>(Wo, wob, w4);
    }

    // 2) Q/K/V projections (WMMA GEMM)
    dim3 gg(M_ROWS / 128, D_DIM / 128);   // 64 x 8
    wmma_gemm<<<gg, 256, 0, stream>>>(xb, wqb, Qb,  D_DIM, 0);
    wmma_gemm<<<gg, 256, 0, stream>>>(xb, wkb, Kb,  D_DIM, 0);
    wmma_gemm<<<gg, 256, 0, stream>>>(xb, wvb, VTb, D_DIM, 1);

    // 3) flash attention
    flash_attn<<<dim3(S_LEN / 64, H_NUM, 4), 128, 0, stream>>>(Qb, Kb, VTb, At, causal);

    // 4) output projection -> f32 d_out
    wmma_gemm<<<gg, 256, 0, stream>>>(At, wob, (float*)d_out, D_DIM, 2);
}
